// ConvAttn_61503931678811
// MI455X (gfx1250) — matrix operand order repaired and checked by reading, not compiled, not run on hardware
//
#include <hip/hip_runtime.h>
#include <hip/hip_fp16.h>

typedef _Float16 v16h __attribute__((ext_vector_type(16)));
typedef _Float16 v8h  __attribute__((ext_vector_type(8)));
typedef float    v8f  __attribute__((ext_vector_type(8)));

union HV16 { v16h v; v8h h[2]; };

#define WMMA_F16(a, b, c) \
  __builtin_amdgcn_wmma_f32_16x16x32_f16(false, (a), false, (b), (short)0, (c), false, false)

// B=2, H=16, L=2048, DK=64, C=H*DK=1024, KER=3
#define Lq   2048
#define Cc   1024
#define NB   2
#define NH   16
#define DKq  64

// --- CDNA5 async global->LDS copy (ASYNCcnt-tracked), inline asm: portable
// across ROCm7.2 / amdgpu-toolchain (bypasses arity-divergent builtins).
__device__ __forceinline__ void async_copy_b128(unsigned lds_addr, const void* gaddr)
{
    asm volatile("global_load_async_to_lds_b128 %0, %1, off"
                 :: "v"(lds_addr), "v"(gaddr) : "memory");
}
__device__ __forceinline__ void async_wait_le16()
{
    asm volatile("s_wait_asynccnt 0x10" ::: "memory");
}
__device__ __forceinline__ void async_wait_0()
{
    asm volatile("s_wait_asynccnt 0x0" ::: "memory");
}

// ---------------------------------------------------------------------------
// Kernel 1: conv1d over last dim of raw-reshaped [B][1024][2048] as a GEMM:
//   Y[co,x] = sum_t sum_ci W[co,ci,t] * X[ci, x+t-1]   (zero padded)
// Tile: 64(M=c_out) x 128(N=x) per 256-thread workgroup, 8 waves in 2x4,
// each wave computes a 32x32 sub-tile = 2x2 WMMA f16 tiles, f32 accumulate.
// LDS fills are register-staged + fully unrolled so global loads pipeline.
// ---------------------------------------------------------------------------
__global__ __launch_bounds__(256)
void conv1d_wmma(const float* __restrict__ Q, const float* __restrict__ K,
                 const float* __restrict__ Wq, const float* __restrict__ Wk,
                 _Float16* __restrict__ qc, _Float16* __restrict__ kc)
{
    const int which = blockIdx.z & 1;
    const int b     = blockIdx.z >> 1;
    const float*    X = which ? K  : Q;
    const float*    W = which ? Wk : Wq;
    _Float16*       Y = which ? kc : qc;

    const int Mb = blockIdx.y * 64;     // c_out base
    const int Nb = blockIdx.x * 128;    // x base

    __shared__ _Float16 As[64][32];     // [c_out][k]   4 KB
    __shared__ _Float16 Bs[128][32];    // [x][k] (transposed) 8 KB

    const int tid  = threadIdx.x;
    const int lane = tid & 31;
    const int wave = tid >> 5;
    const int wm   = (wave >> 2) * 32;  // 0 or 32
    const int wn   = (wave & 3) * 32;   // 0..96

    v8f acc[2][2] = {};

    const float* Xb = X + (size_t)b * (Cc * Lq);

    const int mrow = lane & 15;
    const int kbA  = (lane >> 4) * 8;
    const int kbB  = (lane >> 4) * 16;

    for (int t = 0; t < 3; ++t) {
        for (int k0 = 0; k0 < Cc; k0 += 32) {
            // ---- register-staged tile loads (issue all, wait once) ----
            float areg[8];
#pragma unroll
            for (int j = 0; j < 8; ++j) {
                int i = tid + j * 256;
                int m = i >> 5, kk = i & 31;
                areg[j] = W[(size_t)(Mb + m) * (Cc * 3) + (size_t)(k0 + kk) * 3 + t];
            }
            float breg[16];
#pragma unroll
            for (int j = 0; j < 16; ++j) {
                int i = tid + j * 256;
                int kk = i >> 7, x = i & 127;
                int src = Nb + x + t - 1;
                breg[j] = (src >= 0 && src < Lq) ? Xb[(size_t)(k0 + kk) * Lq + src] : 0.0f;
            }
            if (k0 + 32 < Cc)
                __builtin_prefetch(&Xb[(size_t)(k0 + 32) * Lq + Nb], 0, 3);
#pragma unroll
            for (int j = 0; j < 8; ++j) {
                int i = tid + j * 256;
                As[i >> 5][i & 31] = (_Float16)areg[j];
            }
#pragma unroll
            for (int j = 0; j < 16; ++j) {
                int i = tid + j * 256;
                Bs[i & 127][i >> 7] = (_Float16)breg[j];
            }
            __syncthreads();

            // ---- fragments + WMMA ----
            v16h af[2], bf[2];
#pragma unroll
            for (int mi = 0; mi < 2; ++mi) {
                HV16 u;
                const _Float16* ap = &As[wm + mi * 16 + mrow][0];
                u.h[0] = *(const v8h*)(ap + kbB);        // halves 0-7 : K=kbB..kbB+7
                u.h[1] = *(const v8h*)(ap + kbB + 8);    // halves 8-15: K=kbB+8..
                af[mi] = u.v;
            }
#pragma unroll
            for (int ni = 0; ni < 2; ++ni) {
                HV16 u;
                const _Float16* bp = &Bs[wn + ni * 16 + mrow][0];
                u.h[0] = *(const v8h*)(bp + kbB);        // halves 0-7 : K=kbB..kbB+7
                u.h[1] = *(const v8h*)(bp + kbB + 8);    // halves 8-15: K=kbB+8..
                bf[ni] = u.v;
            }
#pragma unroll
            for (int mi = 0; mi < 2; ++mi)
#pragma unroll
                for (int ni = 0; ni < 2; ++ni)
                    acc[mi][ni] = WMMA_F16(af[mi], bf[ni], acc[mi][ni]);

            __syncthreads();
        }
    }

    // epilogue: C layout -> row = r + 8*(lane>>4), col = lane&15
    _Float16* Yb = Y + (size_t)b * (Cc * Lq);
    const int r0 = (lane >> 4) * 8;
    const int c  = lane & 15;
#pragma unroll
    for (int mi = 0; mi < 2; ++mi)
#pragma unroll
        for (int ni = 0; ni < 2; ++ni)
#pragma unroll
            for (int r = 0; r < 8; ++r) {
                int row = Mb + wm + mi * 16 + r0 + r;
                int col = Nb + wn + ni * 16 + c;
                Yb[(size_t)row * Lq + col] = (_Float16)acc[mi][ni][r];
            }
}

// ---------------------------------------------------------------------------
// Kernel 2: context[b,h,q,d] = sum_v V[b,h,v,d]  (einsum sums k & v separately;
// softmax row-sum == 1).  One workgroup per (b,h): reduce then broadcast-write.
// ---------------------------------------------------------------------------
__global__ __launch_bounds__(256)
void vsum_context(const float* __restrict__ V, float* __restrict__ ctx)
{
    const int bh = blockIdx.x;                       // 0..31
    const float* Vb = V   + (size_t)bh * Lq * DKq;
    float*       Cb = ctx + (size_t)bh * Lq * DKq;
    const int tid = threadIdx.x;
    const int d = tid & 63;
    const int p = tid >> 6;                          // 0..3
    float s = 0.f;
    for (int l = p; l < Lq; l += 4) s += Vb[(size_t)l * DKq + d];
    __shared__ float part[4][DKq];
    __shared__ float vs[DKq];
    part[p][d] = s;
    __syncthreads();
    if (tid < DKq)
        vs[tid] = part[0][tid] + part[1][tid] + part[2][tid] + part[3][tid];
    __syncthreads();
    const float v = vs[d];
    for (int l = p; l < Lq; l += 4) Cb[(size_t)l * DKq + d] = v;
}

// ---------------------------------------------------------------------------
// Kernel 3: S = q k^T / 8 per head (2048x2048x64 WMMA GEMM), fused row
// softmax, single write of attn.  Workgroup = 16 rows x full 2048 cols;
// 8 waves, each owning 256 cols = 16 accumulator tiles (128 VGPRs).
// K-rows are streamed into per-wave LDS double buffers with CDNA5
// GLOBAL_LOAD_ASYNC_TO_LDS_B128 (ASYNCcnt), overlapping DMA with WMMA.
// ---------------------------------------------------------------------------
__global__ __launch_bounds__(256)
void attn_wmma(const _Float16* __restrict__ qc, const _Float16* __restrict__ kc,
               float* __restrict__ attn)
{
    const int m0 = blockIdx.x * 16;
    const int h  = blockIdx.y;
    const int b  = blockIdx.z;
    // raw reshape: head h of batch b is a contiguous 2048x64 row-major matrix
    const size_t headoff = ((size_t)b * NH + h) * (size_t)(Lq * DKq);
    const _Float16* qh = qc + headoff;
    const _Float16* kh = kc + headoff;

    const int tid  = threadIdx.x;
    const int lane = tid & 31;
    const int wave = tid >> 5;
    const int n0   = wave * 256;

    const int mrow = lane & 15;
    const int kbA  = (lane >> 4) * 8;
    const int kbB  = (lane >> 4) * 16;

    // per-wave double-buffered K staging: [wave][buf][64 rows][64 d] f16 = 8KB/buf
    __shared__ _Float16 Ksh[8][2][64][DKq];          // 128 KB
    __shared__ float redmax[8][16];
    __shared__ float redsum[8][16];

    // per-lane DMA pattern: 16 x b128 per 64-row chunk (4 rows / instruction)
    const int arow = lane >> 3;                      // 0..3
    const int abyt = (lane & 7) * 16;                // 0..112 within 128B row

    // helper: issue async stage of chunk c into buffer bf (lambda-free, unrolled)
    #define STAGE_CHUNK(c_, bf_)                                                  \
        do {                                                                      \
            const char* gsrc = (const char*)(kh + (size_t)(n0 + (c_) * 64) * DKq);\
            unsigned lbase = (unsigned)(size_t)&Ksh[wave][(bf_)][0][0];           \
            _Pragma("unroll")                                                     \
            for (int i_ = 0; i_ < 16; ++i_) {                                     \
                int off_ = (i_ * 4 + arow) * 128 + abyt;                          \
                async_copy_b128(lbase + off_, gsrc + off_);                       \
            }                                                                     \
        } while (0)

    STAGE_CHUNK(0, 0);

    // A fragments (16 rows, K=64 in two k32 halves) straight from global
    v16h af[2];
#pragma unroll
    for (int ks = 0; ks < 2; ++ks) {
        HV16 ua;
        const _Float16* ap = qh + (size_t)(m0 + mrow) * DKq + ks * 32;
        ua.h[0] = *(const v8h*)(ap + kbB);
        ua.h[1] = *(const v8h*)(ap + kbB + 8);
        af[ks] = ua.v;
    }

    v8f acc[16] = {};

#pragma unroll
    for (int c = 0; c < 4; ++c) {
        if (c < 3) {
            STAGE_CHUNK(c + 1, (c + 1) & 1);
            async_wait_le16();                      // in-order: chunk c has landed
        } else {
            async_wait_0();
        }
        const _Float16* kbuf = &Ksh[wave][c & 1][0][0];
#pragma unroll
        for (int ks = 0; ks < 2; ++ks)
#pragma unroll
            for (int nt = 0; nt < 4; ++nt) {
                HV16 ub;                            // B col n = k-row n (k^T)
                const _Float16* bp = kbuf + (size_t)(nt * 16 + mrow) * DKq + ks * 32;
                ub.h[0] = *(const v8h*)(bp + kbB);
                ub.h[1] = *(const v8h*)(bp + kbB + 8);
                acc[c * 4 + nt] = WMMA_F16(af[ks], ub.v, acc[c * 4 + nt]);
            }
    }
    #undef STAGE_CHUNK

    // scale by 1/sqrt(64)
#pragma unroll
    for (int nt = 0; nt < 16; ++nt)
#pragma unroll
        for (int r = 0; r < 8; ++r) acc[nt][r] *= 0.125f;

    // -------- fused softmax over the 2048-wide row --------
    const int rbase = (lane >> 4) * 8;               // this half-wave owns rows rbase..+7

    float gmax[8];
    {
        float rmax[8];
#pragma unroll
        for (int r = 0; r < 8; ++r) {
            float m = acc[0][r];
#pragma unroll
            for (int nt = 1; nt < 16; ++nt) m = fmaxf(m, acc[nt][r]);
            for (int off = 1; off < 16; off <<= 1)   // reduce across the 16 cols
                m = fmaxf(m, __shfl_xor(m, off, 32));
            rmax[r] = m;
        }
        if ((lane & 15) == 0)
#pragma unroll
            for (int r = 0; r < 8; ++r) redmax[wave][rbase + r] = rmax[r];
    }
    __syncthreads();
#pragma unroll
    for (int r = 0; r < 8; ++r) {
        float m = redmax[0][rbase + r];
#pragma unroll
        for (int w = 1; w < 8; ++w) m = fmaxf(m, redmax[w][rbase + r]);
        gmax[r] = m;
    }

    float rsum[8] = {0.f, 0.f, 0.f, 0.f, 0.f, 0.f, 0.f, 0.f};
#pragma unroll
    for (int nt = 0; nt < 16; ++nt)
#pragma unroll
        for (int r = 0; r < 8; ++r) {
            float e = __expf(acc[nt][r] - gmax[r]);
            acc[nt][r] = e;
            rsum[r] += e;
        }
#pragma unroll
    for (int r = 0; r < 8; ++r) {
        float s = rsum[r];
        for (int off = 1; off < 16; off <<= 1) s += __shfl_xor(s, off, 32);
        rsum[r] = s;
    }
    if ((lane & 15) == 0)
#pragma unroll
        for (int r = 0; r < 8; ++r) redsum[wave][rbase + r] = rsum[r];
    __syncthreads();

    float rinv[8];
#pragma unroll
    for (int r = 0; r < 8; ++r) {
        float s = 0.f;
#pragma unroll
        for (int w = 0; w < 8; ++w) s += redsum[w][rbase + r];
        rinv[r] = 1.0f / s;
    }

    // -------- single write of the 536 MB attn output --------
    float* ab = attn + ((size_t)b * NH + h) * (size_t)Lq * Lq;
    const int cc = lane & 15;
#pragma unroll
    for (int nt = 0; nt < 16; ++nt)
#pragma unroll
        for (int r = 0; r < 8; ++r) {
            int row = m0 + rbase + r;
            int col = n0 + nt * 16 + cc;
            ab[(size_t)row * Lq + col] = acc[nt][r] * rinv[r];
        }
}

// ---------------------------------------------------------------------------
extern "C" void kernel_launch(void* const* d_in, const int* in_sizes, int n_in,
                              void* d_out, int out_size, void* d_ws, size_t ws_size,
                              hipStream_t stream)
{
    (void)in_sizes; (void)n_in; (void)out_size; (void)ws_size;
    const float* Q  = (const float*)d_in[0];
    const float* K  = (const float*)d_in[1];
    const float* V  = (const float*)d_in[2];
    const float* Wq = (const float*)d_in[3];
    const float* Wk = (const float*)d_in[4];

    float* ctx  = (float*)d_out;                                  // [2,16,2048,64]
    float* attn = (float*)d_out + (size_t)NB * NH * Lq * DKq;     // [2,16,2048,2048]

    _Float16* qc = (_Float16*)d_ws;                               // 8 MB
    _Float16* kc = qc + (size_t)NB * Cc * Lq;                     // 8 MB

    // conv(Q,Wq)->qc and conv(K,Wk)->kc as WMMA GEMMs (f16 out)
    conv1d_wmma<<<dim3(Lq / 128, Cc / 64, NB * 2), 256, 0, stream>>>(Q, K, Wq, Wk, qc, kc);
    // context = broadcast column-sum of V
    vsum_context<<<dim3(NB * NH), 256, 0, stream>>>(V, ctx);
    // scores GEMM + fused softmax -> attn
    attn_wmma<<<dim3(Lq / 16, NH, NB), 256, 0, stream>>>(qc, kc, attn);
}
